// KimiMoE_8418135900650
// MI455X (gfx1250) — matrix operand order repaired and checked by reading, not compiled, not run on hardware
//
#include <hip/hip_runtime.h>
#include <stdint.h>

#if defined(__has_builtin)
#  if __has_builtin(__builtin_amdgcn_tensor_load_to_lds) && __has_builtin(__builtin_amdgcn_s_wait_tensorcnt)
#    define HAS_TDM 1
#  else
#    define HAS_TDM 0
#  endif
#else
#  define HAS_TDM 0
#endif

// ---------------- problem constants (match reference) ----------------
constexpr int T_TOK = 8192, HID = 2048, NEXP = 32, TOPK = 8, IDIM = 1024, NSHI = 2048;
constexpr float ROUTE_SCALE = 2.5f;
// ---------------- GEMM tiling ----------------
constexpr int BM = 128, BN = 64, BK = 32;
constexpr int LP = 8, BKP = BK + LP;      // LDS row pad: 8 bf16 = 16B (4 dwords)

typedef __bf16 bf16;
typedef __attribute__((ext_vector_type(16))) __bf16 v16bf;
typedef __attribute__((ext_vector_type(8)))  float  v8f;
typedef __attribute__((ext_vector_type(4)))  unsigned int u32x4;
typedef __attribute__((ext_vector_type(8)))  int i32x8;
typedef __attribute__((ext_vector_type(4)))  int i32x4;

union FragU { uint4 u[2]; v16bf v; };

__device__ __forceinline__ uint16_t bfbits(float f) {
  uint32_t u = __float_as_uint(f);
  u += 0x7FFFu + ((u >> 16) & 1u);        // round-to-nearest-even
  return (uint16_t)(u >> 16);
}
__device__ __forceinline__ bf16 f2bf(float f) {
  uint16_t h = bfbits(f); bf16 b; __builtin_memcpy(&b, &h, 2); return b;
}

#define WMMA_BF16(A, B, C) \
  __builtin_amdgcn_wmma_f32_16x16x32_bf16(false, (A), false, (B), (short)0, (C), false, false)

// ---------------------------------------------------------------------------
// TDM tile load: [BN rows x BK bf16] strided tile -> LDS, hardware-padded so
// each 64B row lands on an 80B (BKP) pitch.  (CDNA5 ISA)
// ---------------------------------------------------------------------------
#if HAS_TDM
__device__ __forceinline__ void tdm_load_tile(uint32_t lds_byte, const bf16* gsrc,
                                              long long row_stride_elems) {
  const uint64_t ga = (uint64_t)(uintptr_t)gsrc;
  u32x4 g0;
  g0[0] = 1u;                                                 // count = 1 (user mode)
  g0[1] = lds_byte;                                           // lds_addr
  g0[2] = (uint32_t)ga;                                       // global_addr lo
  g0[3] = (uint32_t)((ga >> 32) & 0x01FFFFFFu) | (2u << 30);  // addr[56:32] | type=2
  const uint32_t td = 0x40000000u;      // huge tensor dims -> no OOB clipping
  const uint64_t st = (uint64_t)row_stride_elems;
  i32x8 g1;
  g1[0] = (int)((1u << 16)    // data_size = 2 bytes
              | (1u << 20)    // pad_enable
              | (3u << 22)    // pad_interval: every 16 dwords (one 32-elem row)
              | (3u << 25));  // pad_amount: 4 dwords (= LP bf16)
  g1[1] = (int)((td & 0xFFFFu) << 16);                  // tensor_dim0 lo16
  g1[2] = (int)((td >> 16) | ((td & 0xFFFFu) << 16));   // td0 hi16 | tensor_dim1 lo16
  g1[3] = (int)((td >> 16) | ((uint32_t)BK << 16));     // td1 hi16 | tile_dim0 = 32
  g1[4] = (int)(uint32_t)BN;                            // tile_dim1 = 64, tile_dim2 = 0
  g1[5] = (int)(uint32_t)st;                            // tensor_dim0_stride lo32
  g1[6] = (int)((st >> 32) & 0xFFFFu);                  // stride hi16 | dim1_stride lo16=0
  g1[7] = 0;
  const i32x4 z4 = {0, 0, 0, 0};                        // groups 2/3: unused (2D tile)
  const i32x8 z8 = {0, 0, 0, 0, 0, 0, 0, 0};
  __builtin_amdgcn_tensor_load_to_lds(g0, g1, z4, z4, z8, 0);
}
#endif

// Stage one B tile (weights pre-transposed [N][K] bf16): rows n0..n0+63, k..k+31.
__device__ __forceinline__ void stage_b_tile(bf16 (*dst)[BKP], const bf16* __restrict__ gsrc,
                                             int Kdim, int tid) {
#if HAS_TDM
  if (tid < 32)   // wave 0 issues the DMA (EXEC ignored; descriptor is uniform)
    tdm_load_tile((uint32_t)(uintptr_t)&dst[0][0], gsrc, (long long)Kdim);
#else
  for (int q = tid; q < BN * BK / 8; q += 256) {
    const int n = q >> 2, k8 = (q & 3) << 3;
    *(uint4*)&dst[n][k8] = *(const uint4*)(gsrc + (size_t)n * Kdim + k8);
  }
#endif
}

__device__ __forceinline__ void wait_tiles_and_barrier() {
#if HAS_TDM
  __builtin_amdgcn_s_wait_tensorcnt((short)0);   // TDM done (no-op for non-issuing waves)
#endif
  __syncthreads();
}

// ---------------------------------------------------------------------------
// One-shot conversions: x -> bf16 row-major; weights -> bf16 TRANSPOSED [N][K].
// ---------------------------------------------------------------------------
__global__ void __launch_bounds__(256) cvt_x_kernel(const float* __restrict__ src,
                                                    bf16* __restrict__ dst, long long n) {
  const long long i = ((long long)blockIdx.x * 256 + threadIdx.x) * 4;
  if (i < n) {
    const float4 v = *(const float4*)(src + i);
    const uint32_t lo = ((uint32_t)bfbits(v.y) << 16) | bfbits(v.x);
    const uint32_t hi = ((uint32_t)bfbits(v.w) << 16) | bfbits(v.z);
    *(uint2*)(dst + i) = make_uint2(lo, hi);
  }
}

__global__ void __launch_bounds__(256) transpose_cvt_kernel(
    const float* __restrict__ src, bf16* __restrict__ dst, int K, int N,
    size_t sstride, size_t dstride) {
  __shared__ float tile[32][33];
  const float* s = src + (size_t)blockIdx.z * sstride;
  bf16*        d = dst + (size_t)blockIdx.z * dstride;
  const int n0 = blockIdx.x * 32, k0 = blockIdx.y * 32;
  const int tx = threadIdx.x & 31, ty = threadIdx.x >> 5;   // 32 x 8
  #pragma unroll
  for (int r = 0; r < 32; r += 8)
    tile[ty + r][tx] = s[(size_t)(k0 + ty + r) * N + n0 + tx];
  __syncthreads();
  #pragma unroll
  for (int r = 0; r < 32; r += 8)
    d[(size_t)(n0 + ty + r) * K + k0 + tx] = f2bf(tile[tx][ty + r]);
}

// ---------------------------------------------------------------------------
// Router: one wave per token, lane == expert (E == 32 == wave32).
// ---------------------------------------------------------------------------
__global__ void __launch_bounds__(256) router_kernel(
    const float* __restrict__ x, const float* __restrict__ gw,
    const float* __restrict__ gb, int* __restrict__ out_ids,
    int* __restrict__ counts, int* __restrict__ ilist, float* __restrict__ wlist) {
  const int lane = threadIdx.x & 31;
  const int t = blockIdx.x * 8 + (threadIdx.x >> 5);
  const float* xr = x + (size_t)t * HID;
  float acc = 0.f;
  #pragma unroll 4
  for (int h = 0; h < HID; ++h)
    acc = __fmaf_rn(xr[h], gw[h * NEXP + lane], acc);
  const float score = 1.f / (1.f + __expf(-acc));
  float choice = score + gb[lane];
  float sum = 0.f, myw = 0.f;
  int myid = 0;
  #pragma unroll
  for (int k = 0; k < TOPK; ++k) {
    float best = choice; int bi = lane;
    #pragma unroll
    for (int off = 16; off > 0; off >>= 1) {   // wave32 argmax, low-index ties
      float ob = __shfl_xor(best, off, 32);
      int   oi = __shfl_xor(bi,   off, 32);
      if (ob > best || (ob == best && oi < bi)) { best = ob; bi = oi; }
    }
    float s = __shfl(score, bi, 32);
    sum += s;
    if (lane == k)  { myid = bi; myw = s; }
    if (lane == bi) choice = -3.4e38f;
  }
  if (lane < TOPK) {
    const float w = myw * ROUTE_SCALE / (sum + 1e-20f);
    out_ids[(size_t)t * TOPK + lane] = myid;
    const int pos = atomicAdd(&counts[myid], 1);
    ilist[myid * T_TOK + pos] = t;
    wlist[myid * T_TOK + pos] = w;
  }
}

__global__ void scan_kernel(const int* __restrict__ counts, int* __restrict__ hs_off) {
  if (threadIdx.x == 0 && blockIdx.x == 0) {
    int acc = 0;
    for (int e = 0; e < NEXP; ++e) {
      hs_off[e] = acc;
      acc += ((counts[e] + BM - 1) / BM) * BM;
    }
  }
}

// ---------------------------------------------------------------------------
// Fused SwiGLU up-proj: hs[r,n] = w(r) * silu(x_r.Wg_n) * (x_r.Wu_n).
// bf16 WMMA, f32 acc; double-buffered LDS; B tiles staged by the TDM.
// ---------------------------------------------------------------------------
__global__ void __launch_bounds__(256) swiglu_gemm_kernel(
    const bf16* __restrict__ xb, const bf16* __restrict__ BgT0,
    const bf16* __restrict__ BuT0, bf16* __restrict__ hs,
    int Kdim, int N, size_t bstride,
    const int* __restrict__ ilist, const float* __restrict__ wlist,
    const int* __restrict__ counts, const int* __restrict__ hs_off) {
  __shared__ bf16 As [2][BM][BKP];
  __shared__ bf16 Bsg[2][BN][BKP];
  __shared__ bf16 Bsu[2][BN][BKP];

  const int e = blockIdx.z;
  int cnt = 0x7fffffff, hsbase = 0, eT = 0;
  const bf16* BgT = BgT0; const bf16* BuT = BuT0;
  if (ilist) {
    cnt = counts[e]; hsbase = hs_off[e]; eT = e * T_TOK;
    BgT += (size_t)e * bstride; BuT += (size_t)e * bstride;
    if ((int)(blockIdx.x * BM) >= cnt) return;            // uniform early exit
  }
  const int row0 = blockIdx.x * BM, col0 = blockIdx.y * BN;
  const int tid = threadIdx.x, lane = tid & 31, wid = tid >> 5;
  const int wm = wid & 3, wn = wid >> 2;

  // hoisted gather: this thread's two fixed A-copy rows (tokens independent of k)
  const int am = tid >> 2, ak8 = (tid & 3) << 3;
  const bf16* aptr[2];
  #pragma unroll
  for (int s = 0; s < 2; ++s) {
    const int rl = row0 + am + 64 * s;
    if (!ilist)        aptr[s] = xb + (size_t)rl * Kdim;
    else if (rl < cnt) aptr[s] = xb + (size_t)ilist[eT + rl] * Kdim;
    else               aptr[s] = nullptr;                 // padding row -> zeros
  }
  const bf16* bg_t = BgT + (size_t)col0 * Kdim;           // transposed weights [N][K]
  const bf16* bu_t = BuT + (size_t)col0 * Kdim;

  v8f accg[2][2], accu[2][2];
  #pragma unroll
  for (int i = 0; i < 2; ++i)
    #pragma unroll
    for (int j = 0; j < 2; ++j) {
      accg[i][j] = (v8f){0.f,0.f,0.f,0.f,0.f,0.f,0.f,0.f};
      accu[i][j] = (v8f){0.f,0.f,0.f,0.f,0.f,0.f,0.f,0.f};
    }

  // prologue: stage k-tile 0 into buffer 0
  stage_b_tile(Bsg[0], bg_t, Kdim, tid);
  stage_b_tile(Bsu[0], bu_t, Kdim, tid);
  {
    const uint4 z = {0, 0, 0, 0};
    #pragma unroll
    for (int s = 0; s < 2; ++s)
      *(uint4*)&As[0][am + 64 * s][ak8] = aptr[s] ? *(const uint4*)(aptr[s] + ak8) : z;
  }
  wait_tiles_and_barrier();

  const int nsteps = Kdim / BK;
  for (int step = 0; step < nsteps; ++step) {
    const int cur = step & 1, nxt = cur ^ 1;
    if (step + 1 < nsteps) {                              // overlap next-tile staging
      const int kn = (step + 1) * BK;
      stage_b_tile(Bsg[nxt], bg_t + kn, Kdim, tid);
      stage_b_tile(Bsu[nxt], bu_t + kn, Kdim, tid);
      const uint4 z = {0, 0, 0, 0};
      #pragma unroll
      for (int s = 0; s < 2; ++s)
        *(uint4*)&As[nxt][am + 64 * s][ak8] = aptr[s] ? *(const uint4*)(aptr[s] + kn + ak8) : z;
    }
    FragU a[2], bg[2], bu[2];
    const int ar  = wm * 32 + (lane & 15);
    const int akf = (lane >> 4) << 3;                     // A K-base 0/8
    const int bn  = wn * 32 + (lane & 15);
    const int bk0 = (lane >> 4) << 3;                     // B K-base 0/8
    #pragma unroll
    for (int i = 0; i < 2; ++i) {
      a[i].u[0] = *(const uint4*)&As[cur][ar + 16 * i][akf];
      a[i].u[1] = *(const uint4*)&As[cur][ar + 16 * i][16 + akf];
    }
    #pragma unroll
    for (int j = 0; j < 2; ++j) {
      bg[j].u[0] = *(const uint4*)&Bsg[cur][bn + 16 * j][bk0];
      bg[j].u[1] = *(const uint4*)&Bsg[cur][bn + 16 * j][bk0 + 16];
      bu[j].u[0] = *(const uint4*)&Bsu[cur][bn + 16 * j][bk0];
      bu[j].u[1] = *(const uint4*)&Bsu[cur][bn + 16 * j][bk0 + 16];
    }
    #pragma unroll
    for (int i = 0; i < 2; ++i)
      #pragma unroll
      for (int j = 0; j < 2; ++j) {
        accg[i][j] = WMMA_BF16(a[i].v, bg[j].v, accg[i][j]);
        accu[i][j] = WMMA_BF16(a[i].v, bu[j].v, accu[i][j]);
      }
    wait_tiles_and_barrier();
  }

  // epilogue: hs = combine_weight * silu(g) * u  (padding rows -> exact zeros)
  const int mlo = (lane >> 4) << 3, nn = lane & 15;
  #pragma unroll
  for (int i = 0; i < 2; ++i)
    #pragma unroll
    for (int j = 0; j < 2; ++j)
      #pragma unroll
      for (int r = 0; r < 8; ++r) {
        const int rl = row0 + wm * 32 + i * 16 + mlo + r;
        const int ng = col0 + wn * 32 + j * 16 + nn;
        float sc = 1.f;
        if (ilist) sc = (rl < cnt) ? wlist[eT + rl] : 0.f;
        const float g = accg[i][j][r], u = accu[i][j][r];
        const float hv = g * (1.f / (1.f + __expf(-g))) * u * sc;
        hs[(size_t)(hsbase + rl) * N + ng] = f2bf(hv);
      }
}

// ---------------------------------------------------------------------------
// Down-projection: out[token,:] += hs_rows @ Wd (Wd pre-transposed [H][K]).
// ---------------------------------------------------------------------------
__global__ void __launch_bounds__(256) down_gemm_kernel(
    const bf16* __restrict__ hs, const bf16* __restrict__ BwT0,
    float* __restrict__ out, int Kdim, size_t bstride,
    const int* __restrict__ ilist, const int* __restrict__ counts,
    const int* __restrict__ hs_off) {
  __shared__ bf16 As[2][BM][BKP];
  __shared__ bf16 Bs[2][BN][BKP];

  const int e = blockIdx.z;
  int cnt = 0x7fffffff, hsbase = 0, eT = 0;
  const bf16* BwT = BwT0;
  if (ilist) {
    cnt = counts[e]; hsbase = hs_off[e]; eT = e * T_TOK;
    BwT += (size_t)e * bstride;
    if ((int)(blockIdx.x * BM) >= cnt) return;
  }
  const int row0 = blockIdx.x * BM, col0 = blockIdx.y * BN;
  const int tid = threadIdx.x, lane = tid & 31, wid = tid >> 5;
  const int wm = wid & 3, wn = wid >> 2;

  const int am = tid >> 2, ak8 = (tid & 3) << 3;
  const bf16* aptr[2];
  #pragma unroll
  for (int s = 0; s < 2; ++s)                 // hs rows incl. padding are initialized
    aptr[s] = hs + (size_t)(hsbase + row0 + am + 64 * s) * Kdim;
  const bf16* bw_t = BwT + (size_t)col0 * Kdim;

  v8f acc[2][2];
  #pragma unroll
  for (int i = 0; i < 2; ++i)
    #pragma unroll
    for (int j = 0; j < 2; ++j)
      acc[i][j] = (v8f){0.f,0.f,0.f,0.f,0.f,0.f,0.f,0.f};

  stage_b_tile(Bs[0], bw_t, Kdim, tid);
  #pragma unroll
  for (int s = 0; s < 2; ++s)
    *(uint4*)&As[0][am + 64 * s][ak8] = *(const uint4*)(aptr[s] + ak8);
  wait_tiles_and_barrier();

  const int nsteps = Kdim / BK;
  for (int step = 0; step < nsteps; ++step) {
    const int cur = step & 1, nxt = cur ^ 1;
    if (step + 1 < nsteps) {
      const int kn = (step + 1) * BK;
      stage_b_tile(Bs[nxt], bw_t + kn, Kdim, tid);
      #pragma unroll
      for (int s = 0; s < 2; ++s)
        *(uint4*)&As[nxt][am + 64 * s][ak8] = *(const uint4*)(aptr[s] + kn + ak8);
    }
    FragU a[2], b[2];
    const int ar  = wm * 32 + (lane & 15);
    const int akf = (lane >> 4) << 3;
    const int bn  = wn * 32 + (lane & 15);
    const int bk0 = (lane >> 4) << 3;
    #pragma unroll
    for (int i = 0; i < 2; ++i) {
      a[i].u[0] = *(const uint4*)&As[cur][ar + 16 * i][akf];
      a[i].u[1] = *(const uint4*)&As[cur][ar + 16 * i][16 + akf];
    }
    #pragma unroll
    for (int j = 0; j < 2; ++j) {
      b[j].u[0] = *(const uint4*)&Bs[cur][bn + 16 * j][bk0];
      b[j].u[1] = *(const uint4*)&Bs[cur][bn + 16 * j][bk0 + 16];
    }
    #pragma unroll
    for (int i = 0; i < 2; ++i)
      #pragma unroll
      for (int j = 0; j < 2; ++j)
        acc[i][j] = WMMA_BF16(a[i].v, b[j].v, acc[i][j]);
    wait_tiles_and_barrier();
  }

  const int mlo = (lane >> 4) << 3, nn = lane & 15;
  #pragma unroll
  for (int i = 0; i < 2; ++i)
    #pragma unroll
    for (int j = 0; j < 2; ++j)
      #pragma unroll
      for (int r = 0; r < 8; ++r) {
        const int rl = row0 + wm * 32 + i * 16 + mlo + r;
        const int ng = col0 + wn * 32 + j * 16 + nn;
        const float v = acc[i][j][r];
        if (ilist) {
          if (rl < cnt) {
            const int tok = ilist[eT + rl];
            atomicAdd(out + (size_t)tok * HID + ng, v);
          }
        } else {
          atomicAdd(out + (size_t)rl * HID + ng, v);
        }
      }
}

// ---------------------------------------------------------------------------
extern "C" void kernel_launch(void* const* d_in, const int* in_sizes, int n_in,
                              void* d_out, int out_size, void* d_ws, size_t ws_size,
                              hipStream_t stream) {
  (void)in_sizes; (void)n_in; (void)out_size; (void)ws_size;
  const float* x      = (const float*)d_in[0];
  const float* gate_w = (const float*)d_in[1];
  const float* gate_b = (const float*)d_in[2];
  const float* w_gate = (const float*)d_in[3];
  const float* w_up   = (const float*)d_in[4];
  const float* w_down = (const float*)d_in[5];
  const float* sg_w   = (const float*)d_in[6];
  const float* su_w   = (const float*)d_in[7];
  const float* sd_w   = (const float*)d_in[8];

  float* out     = (float*)d_out;
  int*   out_ids = (int*)(out + (size_t)T_TOK * HID);

  // workspace layout (~640 MB)
  char* ws = (char*)d_ws;
  size_t off = 0;
  auto carve = [&](size_t bytes) { char* p = ws + off; off += (bytes + 255) & ~(size_t)255; return p; };
  int*   counts = (int*)  carve(NEXP * 4);
  int*   hs_off = (int*)  carve(NEXP * 4);
  int*   ilist  = (int*)  carve((size_t)NEXP * T_TOK * 4);
  float* wlist  = (float*)carve((size_t)NEXP * T_TOK * 4);
  bf16*  xb     = (bf16*) carve((size_t)T_TOK * HID * 2);
  bf16*  hs_sh  = (bf16*) carve((size_t)T_TOK * NSHI * 2);
  bf16*  hs_ex  = (bf16*) carve((size_t)(T_TOK * TOPK + NEXP * BM) * IDIM * 2);
  bf16*  sgT    = (bf16*) carve((size_t)NSHI * HID * 2);
  bf16*  suT    = (bf16*) carve((size_t)NSHI * HID * 2);
  bf16*  sdT    = (bf16*) carve((size_t)HID * NSHI * 2);
  bf16*  wgT    = (bf16*) carve((size_t)NEXP * IDIM * HID * 2);
  bf16*  wuT    = (bf16*) carve((size_t)NEXP * IDIM * HID * 2);
  bf16*  wdT    = (bf16*) carve((size_t)NEXP * HID * IDIM * 2);

  (void)hipMemsetAsync(counts, 0, 256, stream);
  (void)hipMemsetAsync(out, 0, (size_t)T_TOK * HID * sizeof(float), stream);

  // one-shot bf16 conversion (+ weight transpose to [N][K])
  const long long nx = (long long)T_TOK * HID;
  cvt_x_kernel<<<(uint32_t)(nx / 4 / 256), 256, 0, stream>>>(x, xb, nx);
  transpose_cvt_kernel<<<dim3(NSHI / 32, HID / 32, 1), 256, 0, stream>>>(
      sg_w, sgT, HID, NSHI, 0, 0);
  transpose_cvt_kernel<<<dim3(NSHI / 32, HID / 32, 1), 256, 0, stream>>>(
      su_w, suT, HID, NSHI, 0, 0);
  transpose_cvt_kernel<<<dim3(HID / 32, NSHI / 32, 1), 256, 0, stream>>>(
      sd_w, sdT, NSHI, HID, 0, 0);
  transpose_cvt_kernel<<<dim3(IDIM / 32, HID / 32, NEXP), 256, 0, stream>>>(
      w_gate, wgT, HID, IDIM, (size_t)HID * IDIM, (size_t)HID * IDIM);
  transpose_cvt_kernel<<<dim3(IDIM / 32, HID / 32, NEXP), 256, 0, stream>>>(
      w_up, wuT, HID, IDIM, (size_t)HID * IDIM, (size_t)HID * IDIM);
  transpose_cvt_kernel<<<dim3(HID / 32, IDIM / 32, NEXP), 256, 0, stream>>>(
      w_down, wdT, IDIM, HID, (size_t)IDIM * HID, (size_t)IDIM * HID);

  router_kernel<<<T_TOK / 8, 256, 0, stream>>>(x, gate_w, gate_b, out_ids,
                                               counts, ilist, wlist);
  scan_kernel<<<1, 32, 0, stream>>>(counts, hs_off);

  // shared expert (identity rows, scale 1)
  swiglu_gemm_kernel<<<dim3(T_TOK / BM, NSHI / BN, 1), 256, 0, stream>>>(
      xb, sgT, suT, hs_sh, HID, NSHI, 0, nullptr, nullptr, nullptr, nullptr);
  down_gemm_kernel<<<dim3(T_TOK / BM, HID / BN, 1), 256, 0, stream>>>(
      hs_sh, sdT, out, NSHI, 0, nullptr, nullptr, nullptr);

  // routed experts (grouped GEMM over gathered token lists)
  swiglu_gemm_kernel<<<dim3(T_TOK / BM, IDIM / BN, NEXP), 256, 0, stream>>>(
      xb, wgT, wuT, hs_ex, HID, IDIM, (size_t)IDIM * HID,
      ilist, wlist, counts, hs_off);
  down_gemm_kernel<<<dim3(T_TOK / BM, HID / BN, NEXP), 256, 0, stream>>>(
      hs_ex, wdT, out, IDIM, (size_t)HID * IDIM, ilist, counts, hs_off);
}
